// RBFLayer_1915555414707
// MI455X (gfx1250) — matrix operand order repaired and checked by reading, not compiled, not run on hardware
//
#include <hip/hip_runtime.h>

// ---------------------------------------------------------------------------
// RBF layer: out[B,K] = exp(-(||x_i||^2 + ||c_j||^2 - 2 x_i.c_j) / (2 s_j^2))
// B=32768, D=256, K=1024.  Cross term via bf16 WMMA (fp32 accum), fp32 epilogue.
// ---------------------------------------------------------------------------

typedef __attribute__((ext_vector_type(16))) __bf16 v16bf;
typedef __attribute__((ext_vector_type(8)))  __bf16 v8bf;
typedef __attribute__((ext_vector_type(4)))  __bf16 v4bf;
typedef __attribute__((ext_vector_type(8)))  float  v8f;

static constexpr int Bsz = 32768;   // rows of x
static constexpr int Dsz = 256;     // feature dim
static constexpr int Ksz = 1024;    // number of centers

static constexpr int TILE_M = 128;
static constexpr int TILE_N = 128;
static constexpr int TILE_K = 32;

__device__ __forceinline__ __bf16 tobf(float f) { return (__bf16)f; }

// ---------------- per-row squared-norm reductions --------------------------

__global__ __launch_bounds__(256)
void rbf_x2_kernel(const float* __restrict__ src, float* __restrict__ dst, int nrows)
{
    int row  = blockIdx.x * 8 + (threadIdx.x >> 5);
    int lane = threadIdx.x & 31;
    if (row >= nrows) return;
    const float4* p = (const float4*)(src + (size_t)row * Dsz);
    float4 a = p[lane * 2 + 0];
    float4 b = p[lane * 2 + 1];
    float s = a.x*a.x + a.y*a.y + a.z*a.z + a.w*a.w
            + b.x*b.x + b.y*b.y + b.z*b.z + b.w*b.w;
#pragma unroll
    for (int m = 16; m > 0; m >>= 1) s += __shfl_xor(s, m, 32);
    if (lane == 0) dst[row] = s;
}

__global__ __launch_bounds__(256)
void rbf_c2_kernel(const float* __restrict__ centers, const float* __restrict__ sigmas,
                   float2* __restrict__ dst, int nrows)
{
    int row  = blockIdx.x * 8 + (threadIdx.x >> 5);
    int lane = threadIdx.x & 31;
    if (row >= nrows) return;
    const float4* p = (const float4*)(centers + (size_t)row * Dsz);
    float4 a = p[lane * 2 + 0];
    float4 b = p[lane * 2 + 1];
    float s = a.x*a.x + a.y*a.y + a.z*a.z + a.w*a.w
            + b.x*b.x + b.y*b.y + b.z*b.z + b.w*b.w;
#pragma unroll
    for (int m = 16; m > 0; m >>= 1) s += __shfl_xor(s, m, 32);
    if (lane == 0) {
        float sg = sigmas[row];
        float2 v; v.x = s; v.y = 1.0f / (2.0f * sg * sg);
        dst[row] = v;
    }
}

// ---------------- main fused GEMM + RBF epilogue ---------------------------

union AFrag { v16bf v; v8bf h2[2]; };

__global__ __launch_bounds__(256)
void rbf_gemm_kernel(const float* __restrict__ X, const float* __restrict__ C,
                     const float* __restrict__ x2, const float2* __restrict__ c2sc,
                     float* __restrict__ out)
{
    __shared__ __align__(16) __bf16 As[TILE_M * TILE_K];   // 8 KB
    __shared__ __align__(16) __bf16 Bs[TILE_N * TILE_K];   // 8 KB
    __shared__ float x2s[TILE_M];
    __shared__ float c2s[TILE_N];
    __shared__ float scs[TILE_N];

    const int tid = threadIdx.x;
    const int bm  = blockIdx.x >> 3;          // 0..255  (M block)
    const int bn  = blockIdx.x & 7;           // 0..7    (N block, fastest -> L2 reuse of x tile)
    const int m0  = bm * TILE_M;
    const int n0  = bn * TILE_N;

    // stage x2 / c2 / scale for this tile
    if (tid < TILE_M) {
        x2s[tid] = x2[m0 + tid];
    } else {
        int t = tid - TILE_M;
        float2 v = c2sc[n0 + t];
        c2s[t] = v.x;
        scs[t] = v.y;
    }

    // global-load assignment: 128 rows x 8 float4 per k-step, 256 threads x 4
    const int frow = tid >> 3;                // 0..31
    const int fcol = tid & 7;                 // 0..7 (float4 within 32-col slab)
    const float4* Xg = (const float4*)X;      // row stride = 64 float4
    const float4* Cg = (const float4*)C;

    float4 ra[4], rb[4];
    auto gload = [&](int kk) {
        const int base = kk * 8 + fcol;
#pragma unroll
        for (int i = 0; i < 4; ++i) {
            const int r = frow + i * 32;
            ra[i] = Xg[(size_t)(m0 + r) * 64 + base];
            rb[i] = Cg[(size_t)(n0 + r) * 64 + base];
        }
    };
    auto stolds = [&]() {
#pragma unroll
        for (int i = 0; i < 4; ++i) {
            const int r = frow + i * 32;
            v4bf a = { tobf(ra[i].x), tobf(ra[i].y), tobf(ra[i].z), tobf(ra[i].w) };
            v4bf b = { tobf(rb[i].x), tobf(rb[i].y), tobf(rb[i].z), tobf(rb[i].w) };
            ((v4bf*)As)[r * 8 + fcol] = a;
            ((v4bf*)Bs)[r * 8 + fcol] = b;
        }
    };

    // wave -> 64x32 sub-tile (2x4 wave grid), 4x2 WMMA tiles each
    const int lane = tid & 31;
    const int wv   = tid >> 5;
    const int wm   = (wv >> 2) * 64;          // 0 / 64
    const int wn   = (wv & 3)  * 32;          // 0 / 32 / 64 / 96
    const int h    = lane >> 4;               // lane half (selects K sub-block)
    const int ml   = lane & 15;               // row (A) / col (B) within tile

    v8f acc[4][2];
#pragma unroll
    for (int mt = 0; mt < 4; ++mt)
#pragma unroll
        for (int nt = 0; nt < 2; ++nt)
            acc[mt][nt] = (v8f){0.f,0.f,0.f,0.f,0.f,0.f,0.f,0.f};

    gload(0);
#pragma unroll 1
    for (int kk = 0; kk < Dsz / TILE_K; ++kk) {
        stolds();                              // fp32 regs -> bf16 LDS
        __syncthreads();
        if (kk < Dsz / TILE_K - 1) gload(kk + 1);  // overlap next tile with WMMA

        AFrag a[4], b[2];
#pragma unroll
        for (int mt = 0; mt < 4; ++mt) {
            const int row = wm + mt * 16 + ml;
            // 16-bit A layout: lanes 0-15 -> K[0..7]+K[16..23], lanes 16-31 -> K[8..15]+K[24..31]
            a[mt].h2[0] = *(const v8bf*)&As[row * TILE_K + 8 * h];
            a[mt].h2[1] = *(const v8bf*)&As[row * TILE_K + 16 + 8 * h];
        }
#pragma unroll
        for (int nt = 0; nt < 2; ++nt) {
            const int col = wn + nt * 16 + ml;
            // 16-bit B layout: lanes 0-15 -> K[0..7]+K[16..23], lanes 16-31 -> K[8..15]+K[24..31] of column `col`
            b[nt].h2[0] = *(const v8bf*)&Bs[col * TILE_K + 8 * h];
            b[nt].h2[1] = *(const v8bf*)&Bs[col * TILE_K + 16 + 8 * h];
        }
#pragma unroll
        for (int mt = 0; mt < 4; ++mt)
#pragma unroll
            for (int nt = 0; nt < 2; ++nt)
                acc[mt][nt] = __builtin_amdgcn_wmma_f32_16x16x32_bf16(
                    false, a[mt].v, false, b[nt].v,
                    (short)0, acc[mt][nt], false, false);
        __syncthreads();
    }

    // fused epilogue: exp(-(x2 + c2 - 2*cross) * inv2s2)
#pragma unroll
    for (int mt = 0; mt < 4; ++mt) {
#pragma unroll
        for (int nt = 0; nt < 2; ++nt) {
            const int lc = wn + nt * 16 + ml;       // local col
            const int j  = n0 + lc;
            const float cc = c2s[lc];
            const float sc = scs[lc];
#pragma unroll
            for (int r = 0; r < 8; ++r) {
                const int lr = wm + mt * 16 + r + 8 * h;   // C layout: VGPR r -> row r + 8*half
                const int i  = m0 + lr;
                const float cross = acc[mt][nt][r];
                const float d2 = x2s[lr] + cc - 2.0f * cross;
                out[(size_t)i * Ksz + j] = __expf(-d2 * sc);
            }
        }
    }
}

// ---------------------------------------------------------------------------

extern "C" void kernel_launch(void* const* d_in, const int* in_sizes, int n_in,
                              void* d_out, int out_size, void* d_ws, size_t ws_size,
                              hipStream_t stream) {
    (void)in_sizes; (void)n_in; (void)out_size; (void)ws_size;

    const float* x       = (const float*)d_in[0];
    const float* centers = (const float*)d_in[1];
    const float* sigmas  = (const float*)d_in[2];
    float*       out     = (float*)d_out;

    // workspace layout: [ c2sc: K * float2 (8KB) | x2: B * float (128KB) ]
    float2* c2sc = (float2*)d_ws;
    float*  x2   = (float*)((char*)d_ws + Ksz * sizeof(float2));

    rbf_x2_kernel<<<Bsz / 8, 256, 0, stream>>>(x, x2, Bsz);
    rbf_c2_kernel<<<Ksz / 8, 256, 0, stream>>>(centers, sigmas, c2sc, Ksz);

    const int grid = (Bsz / TILE_M) * (Ksz / TILE_N);   // 256 * 8 = 2048 blocks
    rbf_gemm_kernel<<<grid, 256, 0, stream>>>(x, centers, x2, c2sc, out);
}
